// IndexPropagationQuantize1D_80985903334283
// MI455X (gfx1250) — matrix operand order repaired and checked by reading, not compiled, not run on hardware
//
#include <hip/hip_runtime.h>
#include <hip/hip_bf16.h>

// Problem constants (from reference)
#define B_    16
#define D_    512
#define L_    3000
#define NE_   1024
#define BETA_ 0.25f
#define NT_   ((L_ + 15) / 16)        // 188 column tiles
#define NBLK_ (B_ * NT_)              // 3008 workgroups
#define ZQ_N_ (B_ * D_ * L_)          // 24,576,000
#define IND_OFF_  ZQ_N_
#define DIFF_OFF_ (ZQ_N_ + B_ * L_)
#define INV_CNT_  (1.0f / (float)ZQ_N_)

typedef __bf16 bf16;
typedef __attribute__((ext_vector_type(8)))  bf16  v8bf;
typedef __attribute__((ext_vector_type(16))) bf16  v16bf;
typedef __attribute__((ext_vector_type(8)))  float v8f;

typedef __attribute__((address_space(1))) int* gint_p;   // global
typedef __attribute__((address_space(3))) int* lint_p;   // LDS

#if defined(__AMDGCN__) && __has_builtin(__builtin_amdgcn_global_load_async_to_lds_b32) && \
    __has_builtin(__builtin_amdgcn_s_wait_asynccnt)
#define USE_ASYNC_LDS 1
#else
#define USE_ASYNC_LDS 0
#endif

__device__ __forceinline__ v16bf cat8(v8bf lo, v8bf hi) {
  return __builtin_shufflevector(lo, hi, 0,1,2,3,4,5,6,7,8,9,10,11,12,13,14,15);
}

// ---------------------------------------------------------------------------
// Kernel 0: convert fp32 codebook to bf16, row-major (Wb, for GEMM1 A) and
// transposed (WbT, for GEMM2 A). 1 MB each, lives in L2 for the whole run.
// ---------------------------------------------------------------------------
__global__ void convert_w(const float* __restrict__ W,
                          bf16* __restrict__ Wb, bf16* __restrict__ WbT) {
  int i = blockIdx.x * blockDim.x + threadIdx.x;
  if (i >= NE_ * D_) return;
  int n = i >> 9;          // / D_
  int d = i & (D_ - 1);
  float w = W[i];
  bf16 h = (bf16)w;
  Wb[i] = h;
  WbT[(size_t)d * NE_ + n] = h;
}

// ---------------------------------------------------------------------------
// Kernel 1 (fused): per (batch, 16-column tile):
//   logits(1024x16) -> softmax/argmax over codes -> z_q_soft(512x16)
//   -> z_q / ind / loss-partial.  16 waves, bf16 WMMA, fp32 accumulate.
// ---------------------------------------------------------------------------
__global__ __launch_bounds__(512) void vq_fused(
    const float* __restrict__ z, const float* __restrict__ W,
    const bf16* __restrict__ Wb, const bf16* __restrict__ WbT,
    float* __restrict__ out, float* __restrict__ partials) {

  __shared__ __align__(16) bf16  sS[16][NE_];   // soft weights, B-operand layout
  __shared__ __align__(16) float sZf[16][32];   // staged z tile (transposed, fp32)
  __shared__ float    sRedV[32][16];
  __shared__ unsigned sRedI[32][16];
  __shared__ float    sColMax[16];
  __shared__ float    sColSum[16];
  __shared__ unsigned sColArg[16];
  __shared__ float    sLoss[512];

  const int tid  = threadIdx.x;
  const int wid  = tid >> 5;        // wave 0..15
  const int lane = tid & 31;
  const int half = lane >> 4;       // 0: lanes 0-15, 1: lanes 16-31
  const int mrow = lane & 15;       // row (A) / column (B,C,D) within tile

  const int blk = blockIdx.x;
  const int b   = blk / NT_;
  const int lt  = blk - b * NT_;
  const int l0  = lt * 16;

  const int skk = tid >> 4;         // staged k within tile: 0..31
  const int snn = tid & 15;         // staged column: 0..15
  const bool svalid = (l0 + snn) < L_;

  // Pre-zero tail columns once; they are never (async-)written afterwards,
  // so they stay zero across all k-steps (no per-iteration zeroing race).
  if (!svalid) sZf[snn][skk] = 0.0f;
  __syncthreads();

  // ---------------- Phase 1: logits tile (NE x 16) --------------------------
  v8f acc[4] = {};
  for (int k0 = 0; k0 < D_; k0 += 32) {
    // stage z[k0..k0+31][l0..l0+15] fp32, transposed [n][k]
    if (svalid) {
      const float* g = &z[((size_t)(b * D_ + (k0 + skk))) * L_ + (l0 + snn)];
#if USE_ASYNC_LDS
      // CDNA5 async copy: global -> LDS, no VGPR round trip (ASYNCcnt)
      __builtin_amdgcn_global_load_async_to_lds_b32(
          (gint_p)g, (lint_p)&sZf[snn][skk], 0, 0);
#else
      sZf[snn][skk] = *g;
#endif
    }
#if USE_ASYNC_LDS
    __builtin_amdgcn_s_wait_asynccnt(0);
#endif
    __syncthreads();

    // B fragment (z): column n = mrow, K = half*8..+7 and 16+half*8..+7, cvt->bf16
    {
      const float* zc = &sZf[mrow][half * 8];
      v8bf blo, bhi;
      #pragma unroll
      for (int i = 0; i < 8; ++i) {
        blo[i] = (bf16)zc[i];
        bhi[i] = (bf16)zc[16 + i];
      }
      v16bf bf = cat8(blo, bhi);

      #pragma unroll
      for (int mi = 0; mi < 4; ++mi) {
        int m = wid * 64 + mi * 16 + mrow;          // code row, 16 waves * 64 = 1024
        const bf16* row = Wb + (size_t)m * D_ + k0;
        v8bf alo = *reinterpret_cast<const v8bf*>(row + half * 8);      // K k0..+7 / +8..15
        v8bf ahi = *reinterpret_cast<const v8bf*>(row + 16 + half * 8); // K +16..23 / +24..31
        __builtin_prefetch(row + 32, 0, 3);         // next K chunk, near-cache locality
        acc[mi] = __builtin_amdgcn_wmma_f32_16x16x32_bf16(
            false, cat8(alo, ahi), false, bf, (short)0, acc[mi], false, false);
      }
    }
    __syncthreads();
  }

  // ---------------- Softmax + argmax over the 1024 codes --------------------
  // C layout: VGPR r, lane -> row m0 + r + 8*half, col n = mrow
  float vmax = -__builtin_inff();
  unsigned varg = 0;
  #pragma unroll
  for (int mi = 0; mi < 4; ++mi)
    #pragma unroll
    for (int r = 0; r < 8; ++r) {
      float v = acc[mi][r];
      unsigned m = (unsigned)(wid * 64 + mi * 16 + half * 8 + r);
      if (v > vmax) { vmax = v; varg = m; }        // ascending m -> first max wins
    }
  int c = wid * 2 + half;                          // contributor 0..31 per column
  sRedV[c][mrow] = vmax;
  sRedI[c][mrow] = varg;
  __syncthreads();
  if (tid < 16) {
    float mx = sRedV[0][tid]; unsigned ag = sRedI[0][tid];
    #pragma unroll 4
    for (int i = 1; i < 32; ++i) {
      float v = sRedV[i][tid]; unsigned a = sRedI[i][tid];
      if (v > mx || (v == mx && a < ag)) { mx = v; ag = a; }
    }
    sColMax[tid] = mx; sColArg[tid] = ag;
    int l = l0 + tid;
    if (l < L_) out[IND_OFF_ + (size_t)b * L_ + l] = (float)ag;  // ind output
  }
  __syncthreads();

  float cmax = sColMax[mrow];
  float esum = 0.0f;
  #pragma unroll
  for (int mi = 0; mi < 4; ++mi)
    #pragma unroll
    for (int r = 0; r < 8; ++r)
      esum += __expf(acc[mi][r] - cmax);
  sRedV[c][mrow] = esum;
  __syncthreads();
  if (tid < 16) {
    float s = 0.0f;
    #pragma unroll 4
    for (int i = 0; i < 32; ++i) s += sRedV[i][tid];
    sColSum[tid] = s;
  }
  __syncthreads();
  float rinv = 1.0f / sColSum[mrow];

  // write soft weights (bf16) into sS[n][k], already in B-operand order
  #pragma unroll
  for (int mi = 0; mi < 4; ++mi) {
    v8bf sv;
    #pragma unroll
    for (int r = 0; r < 8; ++r)
      sv[r] = (bf16)(__expf(acc[mi][r] - cmax) * rinv);
    *reinterpret_cast<v8bf*>(&sS[mrow][wid * 64 + mi * 16 + half * 8]) = sv;
  }
  __syncthreads();

  // ---------------- Phase 2: z_q_soft tile (D x 16), K = NE -----------------
  v8f acc2[2] = {};
  #pragma unroll 8
  for (int k0 = 0; k0 < NE_; k0 += 32) {
    v8bf blo = *reinterpret_cast<const v8bf*>(&sS[mrow][k0 + half * 8]);
    v8bf bhi = *reinterpret_cast<const v8bf*>(&sS[mrow][k0 + 16 + half * 8]);
    v16bf bf = cat8(blo, bhi);
    #pragma unroll
    for (int mi = 0; mi < 2; ++mi) {
      int m = wid * 32 + mi * 16 + mrow;           // d row, 16 waves * 32 = 512
      const bf16* row = WbT + (size_t)m * NE_ + k0;
      v8bf alo = *reinterpret_cast<const v8bf*>(row + half * 8);
      v8bf ahi = *reinterpret_cast<const v8bf*>(row + 16 + half * 8);
      acc2[mi] = __builtin_amdgcn_wmma_f32_16x16x32_bf16(
          false, cat8(alo, ahi), false, bf, (short)0, acc2[mi], false, false);
    }
  }

  // ---------------- Epilogue: z_q, loss partial ------------------------------
  float lpart = 0.0f;
  int l = l0 + mrow;
  unsigned idx = sColArg[mrow];
  bool valid = (l < L_);
  #pragma unroll
  for (int mi = 0; mi < 2; ++mi) {
    #pragma unroll
    for (int r = 0; r < 8; ++r) {
      int m = wid * 32 + mi * 16 + half * 8 + r;
      float soft = acc2[mi][r];
      if (valid) {
        float hard = W[(size_t)idx * D_ + m];              // fp32 gather of code row
        float zq   = (hard + soft) - soft;                 // straight-through value
        size_t off = ((size_t)(b * D_ + m)) * L_ + l;
        out[off] = zq;
        float zf = z[off];
        float d1 = zq - zf;
        float d2 = hard - zf;
        lpart += d1 * d1 + (1.0f + BETA_) * d2 * d2;
      }
    }
  }
  sLoss[tid] = lpart;
  __syncthreads();
  #pragma unroll
  for (int s = 256; s > 0; s >>= 1) {
    if (tid < s) sLoss[tid] += sLoss[tid + s];
    __syncthreads();
  }
  if (tid == 0) partials[blk] = sLoss[0];
}

// ---------------------------------------------------------------------------
// Kernel 2: deterministic final loss reduction
// ---------------------------------------------------------------------------
__global__ void reduce_loss(const float* __restrict__ partials,
                            float* __restrict__ out) {
  __shared__ float s[256];
  float acc = 0.0f;
  for (int i = threadIdx.x; i < NBLK_; i += 256) acc += partials[i];
  s[threadIdx.x] = acc;
  __syncthreads();
  for (int k = 128; k > 0; k >>= 1) {
    if ((int)threadIdx.x < k) s[threadIdx.x] += s[threadIdx.x + k];
    __syncthreads();
  }
  if (threadIdx.x == 0) out[DIFF_OFF_] = s[0] * INV_CNT_;
}

// ---------------------------------------------------------------------------
extern "C" void kernel_launch(void* const* d_in, const int* in_sizes, int n_in,
                              void* d_out, int out_size, void* d_ws, size_t ws_size,
                              hipStream_t stream) {
  const float* z = (const float*)d_in[0];           // (B, D, L) fp32
  const float* W = (const float*)d_in[1];           // (NE, D) fp32
  float* out = (float*)d_out;                       // [z_q | ind | diff]
  char* ws = (char*)d_ws;

  bf16* Wb  = (bf16*)ws;                                            // NE x D   bf16
  bf16* WbT = (bf16*)(ws + (size_t)NE_ * D_ * sizeof(bf16));        // D  x NE  bf16
  float* partials = (float*)(ws + 2u * (size_t)NE_ * D_ * sizeof(bf16)); // NBLK_ floats

  convert_w<<<(NE_ * D_ + 255) / 256, 256, 0, stream>>>(W, Wb, WbT);
  vq_fused<<<NBLK_, 512, 0, stream>>>(z, W, Wb, WbT, out, partials);
  reduce_loss<<<1, 256, 0, stream>>>(partials, out);
}
